// SimpleLSHAttention16_55757265437050
// MI455X (gfx1250) — matrix operand order repaired and checked by reading, not compiled, not run on hardware
//
#include <hip/hip_runtime.h>
#include <hip/hip_fp16.h>
#include <hip/hip_bf16.h>

typedef __attribute__((ext_vector_type(16))) _Float16 v16h;
typedef __attribute__((ext_vector_type(8)))  _Float16 v8h;
typedef __attribute__((ext_vector_type(8)))  float    v8f;

#define SLEN  2048      // sequence length S
#define DDIM  64        // D
#define KPAD  96        // D+1=65 padded to 3 x 32 for WMMA K-chunks
#define NROWS (2*8*2048) // B*H*S
#define NHEAD 16        // B*H

// ---------------------------------------------------------------- init
__global__ void init_ws_kernel(unsigned* maxsq) {
    if (threadIdx.x == 0) *maxsq = 0u;
}

// ------------------------------------------- pass 1: global max ||row||^2
// One wave per row (32 lanes x 2 elems). sumsq >= 0 so uint-bit atomicMax
// is order-equivalent to float max.
__global__ void rownorm_max_kernel(const float* __restrict__ qk,
                                   unsigned* __restrict__ maxsq) {
    int row  = blockIdx.x * 8 + (threadIdx.x >> 5);
    int lane = threadIdx.x & 31;
    float2 v = *(const float2*)(qk + (size_t)row * DDIM + lane * 2);
    float a0 = (float)(_Float16)v.x;
    float a1 = (float)(_Float16)v.y;
    float ss = a0 * a0 + a1 * a1;
    #pragma unroll
    for (int off = 16; off; off >>= 1) ss += __shfl_xor(ss, off, 32);
    if (lane == 0) atomicMax(maxsq, __float_as_uint(ss));
}

// ----------------------- pass 2: build padded qke, padded a, diagonal Q
__global__ void build_qke_kernel(const float* __restrict__ qk,
                                 const _Float16* __restrict__ a,
                                 const unsigned* __restrict__ maxsq,
                                 _Float16* __restrict__ qke,
                                 _Float16* __restrict__ apad,
                                 float* __restrict__ Qd) {
    int row  = blockIdx.x * 8 + (threadIdx.x >> 5);
    int lane = threadIdx.x & 31;
    float invM = 1.0f / sqrtf(__uint_as_float(*maxsq));

    float2 v = *(const float2*)(qk + (size_t)row * DDIM + lane * 2);
    _Float16 h0 = (_Float16)((float)(_Float16)v.x * invM);
    _Float16 h1 = (_Float16)((float)(_Float16)v.y * invM);
    float f0 = (float)h0, f1 = (float)h1;

    float ss = f0 * f0 + f1 * f1;
    #pragma unroll
    for (int off = 16; off; off >>= 1) ss += __shfl_xor(ss, off, 32);
    float c = sqrtf(1.0f - ss);          // NaN propagates like the reference

    const _Float16* arow = a + (size_t)row * (DDIM + 1);
    _Float16 a0 = arow[lane * 2], a1 = arow[lane * 2 + 1];
    float qpart = f0 * (float)a0 + f1 * (float)a1;
    #pragma unroll
    for (int off = 16; off; off >>= 1) qpart += __shfl_xor(qpart, off, 32);

    _Float16* qrow = qke  + (size_t)row * KPAD;
    _Float16* prow = apad + (size_t)row * KPAD;
    qrow[lane * 2]     = h0;  qrow[lane * 2 + 1] = h1;
    prow[lane * 2]     = a0;  prow[lane * 2 + 1] = a1;

    _Float16 alast = arow[DDIM];
    if (lane < 16) {                      // pad region [64,96)
        qrow[DDIM + lane * 2]     = (lane == 0) ? (_Float16)c : (_Float16)0.0f;
        qrow[DDIM + lane * 2 + 1] = (_Float16)0.0f;
        prow[DDIM + lane * 2]     = (lane == 0) ? alast : (_Float16)0.0f;
        prow[DDIM + lane * 2 + 1] = (_Float16)0.0f;
    }
    if (lane == 0) Qd[row] = qpart + c * (float)alast;
}

// --------- pass 3: fused WMMA scores -> top-32 -> bitmap -> fp32 output
// Block = 256 threads (8 waves) = one (bh, 16-row j-tile).
__launch_bounds__(256, 1)
__global__ void score_topk_kernel(const _Float16* __restrict__ qke,
                                  const _Float16* __restrict__ apad,
                                  const float* __restrict__ Qd,
                                  const float* __restrict__ mask,
                                  float* __restrict__ out) {
    extern __shared__ char smem[];
    _Float16* aT = (_Float16*)smem;                            // 16*96*2   = 3072 B
    _Float16* sc = (_Float16*)(smem + 16 * KPAD * 2);          // 16*2048*2 = 65536 B
    unsigned* bm = (unsigned*)(smem + 16 * KPAD * 2 + 16 * SLEN * 2); // 4096 B

    int bh    = blockIdx.x >> 7;       // 128 j-tiles per head
    int jBase = (blockIdx.x & 127) * 16;
    int b     = bh >> 3;               // H = 8
    int tid   = threadIdx.x;
    int lane  = tid & 31;
    int wave  = tid >> 5;

    const _Float16* qkeH = qke  + (size_t)bh * SLEN * KPAD;
    const _Float16* aH   = apad + (size_t)bh * SLEN * KPAD;
    const float*    QH   = Qd   + (size_t)bh * SLEN;
    const float*    mB   = mask + (size_t)b  * SLEN;

    // Stage the 16-row A tile (a rows jBase..jBase+15, KPAD halfs each).
    for (int c = tid; c < 16 * KPAD / 8; c += 256)
        *((v8h*)aT + c) = *((const v8h*)(aH + (size_t)jBase * KPAD) + c);
    for (int c = tid; c < 16 * 64; c += 256) bm[c] = 0;
    __syncthreads();

    // Per-lane WMMA fragment addressing (CDNA5 16x32 f16 A / 32x16 f16 B).
    int m      = lane & 15;
    int kbaseA = (lane < 16) ? 0 : 16;
    int koffB  = (lane < 16) ? 0 : 16;
    int bcol   = lane & 15;

    // A fragments are loop-invariant per wave: build all three K-chunks once.
    v16h afrag[3];
    {
        const _Float16* arow = aT + m * KPAD + kbaseA;
        #pragma unroll
        for (int kc = 0; kc < 3; kc++) {
            v8h lo = *(const v8h*)(arow + kc * 32);
            v8h hi = *(const v8h*)(arow + kc * 32 + 8);
            #pragma unroll
            for (int t = 0; t < 8; t++) { afrag[kc][t] = lo[t]; afrag[kc][8 + t] = hi[t]; }
        }
    }

    for (int chunk = wave; chunk < SLEN / 16; chunk += 8) {
        int iBase = chunk * 16;
        v8f acc = {};
        const _Float16* brow = qkeH + (size_t)(iBase + bcol) * KPAD + koffB;
        #pragma unroll
        for (int kc = 0; kc < 3; kc++) {
            v16h bv = *(const v16h*)(brow + kc * 32);
            acc = __builtin_amdgcn_wmma_f32_16x16x32_f16(
                false, afrag[kc], false, bv, (short)0, acc, false, false);
        }
        int   i  = iBase + bcol;
        float qv = QH[i];
        float mv = mB[i];
        int mrow0 = (lane < 16) ? 0 : 8;
        #pragma unroll
        for (int r = 0; r < 8; r++) {
            float s = acc[r] * qv;        // scores^T[j,i] = Q[i] * P[i,j]
            if (s != s) s = 0.0f;         // NaN -> 0 (reference semantics)
            s += mv;
            sc[(size_t)(mrow0 + r) * SLEN + i] = (_Float16)s;
        }
    }
    __syncthreads();

    // Top-32 per row: wave w owns rows 2w, 2w+1; 32 rounds of wave argmax.
    // Each lane scans with 16B LDS loads (8 halves) -> 8 vector loads/round.
    for (int rr = 0; rr < 2; rr++) {
        int row = wave * 2 + rr;
        _Float16* srow = sc + (size_t)row * SLEN;
        for (int it = 0; it < 32; it++) {
            float best = -3.0e38f; int bidx = 0x7fffffff;
            #pragma unroll
            for (int step = 0; step < SLEN / (32 * 8); step++) {
                int base = (step * 32 + lane) * 8;
                v8h ch = *(const v8h*)(srow + base);
                #pragma unroll
                for (int e = 0; e < 8; e++) {
                    float v = (float)ch[e];
                    if (v > best) { best = v; bidx = base + e; }
                }
            }
            #pragma unroll
            for (int off = 16; off; off >>= 1) {
                float ov = __shfl_xor(best, off, 32);
                int   oi = __shfl_xor(bidx, off, 32);
                if (ov > best || (ov == best && oi < bidx)) { best = ov; bidx = oi; }
            }
            if (lane == 0) {
                srow[bidx] = (_Float16)(-__builtin_huge_valf());
                atomicOr(&bm[row * 64 + (bidx >> 5)], 1u << (bidx & 31));
            }
            __builtin_amdgcn_wave_barrier();
            asm volatile("" ::: "memory");
        }
    }
    __syncthreads();

    // Single-pass fp32 output: -10000 / 0 from bitmap, 16B stores.
    float* outB = out + ((size_t)bh * SLEN + jBase) * SLEN;
    for (int g = tid; g < 16 * (SLEN / 4); g += 256) {
        int row = g >> 9;                 // 512 float4-groups per row
        int col = (g & 511) * 4;
        unsigned w = bm[row * 64 + (col >> 5)];
        float4 val;
        val.x = ((w >> ( col      & 31)) & 1) ? 0.0f : -10000.0f;
        val.y = ((w >> ((col + 1) & 31)) & 1) ? 0.0f : -10000.0f;
        val.z = ((w >> ((col + 2) & 31)) & 1) ? 0.0f : -10000.0f;
        val.w = ((w >> ((col + 3) & 31)) & 1) ? 0.0f : -10000.0f;
        *(float4*)(outB + (size_t)row * SLEN + col) = val;
    }
}

// ---------------------------------------------------------------- launch
extern "C" void kernel_launch(void* const* d_in, const int* in_sizes, int n_in,
                              void* d_out, int out_size, void* d_ws, size_t ws_size,
                              hipStream_t stream) {
    const float*    qk   = (const float*)d_in[0];     // [B,H,S,D] fp32
    const float*    mask = (const float*)d_in[1];     // [B,1,1,S] fp32
    const _Float16* a    = (const _Float16*)d_in[2];  // [B,H,S,D+1] fp16
    // d_in[3] = bucket_size (32), baked into the kernel
    (void)in_sizes; (void)n_in; (void)out_size;

    char* ws = (char*)d_ws;
    const size_t NB = (size_t)NROWS * KPAD * sizeof(_Float16);   // 6.29 MB each
    unsigned* maxsq = (unsigned*)ws;
    _Float16* qke   = (_Float16*)(ws + 256);
    _Float16* apad  = (_Float16*)(ws + 256 + NB);
    float*    Qd    = (float*)   (ws + 256 + 2 * NB);
    float*    out   = (float*)d_out;
    (void)ws_size;

    init_ws_kernel<<<1, 1, 0, stream>>>(maxsq);
    rownorm_max_kernel<<<NROWS / 8, 256, 0, stream>>>(qk, maxsq);
    build_qke_kernel<<<NROWS / 8, 256, 0, stream>>>(qk, a, maxsq, qke, apad, Qd);

    size_t shmem = 16 * KPAD * 2 + 16 * SLEN * 2 + 16 * 64 * 4;  // 72704 B
    hipFuncSetAttribute((const void*)score_topk_kernel,
                        hipFuncAttributeMaxDynamicSharedMemorySize, (int)shmem);
    score_topk_kernel<<<NHEAD * (SLEN / 16), 256, shmem, stream>>>(
        qke, apad, Qd, mask, out);
}
